// TextEncoder_764504179349
// MI455X (gfx1250) — matrix operand order repaired and checked by reading, not compiled, not run on hardware
//
#include <hip/hip_runtime.h>
#include <hip/hip_bf16.h>
#include <stdint.h>

typedef __bf16 bf16;
typedef bf16  v16bf __attribute__((ext_vector_type(16)));
typedef float v8f   __attribute__((ext_vector_type(8)));
typedef unsigned int v4u __attribute__((ext_vector_type(4)));
typedef int v8i __attribute__((ext_vector_type(8)));
typedef int v4i __attribute__((ext_vector_type(4)));

struct __attribute__((aligned(16))) U4 { unsigned int a0, a1, a2, a3; };
struct __attribute__((aligned(16))) F4 { float x, y, z, w; };
union Frag { v16bf v; U4 u[2]; };

#define B_    2048
#define L_    64
#define H_    20
#define DK_   20
#define D_    300
#define HD_   400
#define DPAD  320       // K padding for QKV GEMM (300 -> 320)
#define KT_QKV 10       // 320/32
#define NT_QKV 25       // 400/16
#define CTXPAD 416      // K padding for pooling GEMM (400 -> 416)
#define KT_E   13       // 416/32
#define NT_E   13       // 208/16
#define NPADE  208

#define WPK_ELEMS  (KT_QKV*NT_QKV*32*16)   // 128000 per matrix
#define W1PK_ELEMS (KT_E*NT_E*32*16)       // 86528
#define XBF_BYTES  ((size_t)B_*L_*DPAD*2)  // 83,886,080

// ---- LDS layout (bytes), dynamic shared, 285,184 total (<= 320KB WGP LDS) ----
#define OFF_Q     0          // [64][400] bf16   51200
#define OFF_K     51200      // [64][400] bf16   51200
#define OFF_V     102400     // [64][400] bf16   51200
#define OFF_CTX   153600     // [64][416] bf16   53248
#define OFF_S     206848     // [64][64]  f32    16384
#define OFF_ATTN  223232     // [64][64]  bf16    8192
#define OFF_E     231424     // [64][208] f32    53248
#define OFF_ALPHA 284672     // [64]      f32      256
#define OFF_RS    284928     // [64]      f32      256
#define LDS_BYTES 285184
#define OFF_XS    OFF_S      // x staging [64][320] bf16 = 40960B, overlays S/ATTN/E (phase-1 only)

static __device__ __forceinline__ v8f wmma_bf16(const Frag& a, const Frag& b, v8f c) {
    return __builtin_amdgcn_wmma_f32_16x16x32_bf16(false, a.v, false, b.v, (short)0, c, false, false);
}

// Pack a [300 x 400] f32 weight into WMMA-B fragment order, K padded to 320.
__global__ void pack_qkv(const float* __restrict__ W, bf16* __restrict__ out) {
    int idx = blockIdx.x * 256 + threadIdx.x;
    if (idx >= WPK_ELEMS) return;
    int h    = idx & 15;
    int lane = (idx >> 4) & 31;
    int tile = idx >> 9;                 // kt*NT_QKV + tn
    int tn = tile % NT_QKV;
    int kt = tile / NT_QKV;
    int k  = kt * 32 + (h & 7) + 8 * (lane >> 4) + 16 * (h >> 3);
    int n  = tn * 16 + (lane & 15);
    out[idx] = (k < D_) ? (bf16)W[k * HD_ + n] : (bf16)0.0f;
}

// Pack W1 [400 x 200] f32 into WMMA-B fragment order, K pad 416, N pad 208.
__global__ void pack_w1(const float* __restrict__ W1, bf16* __restrict__ out) {
    int idx = blockIdx.x * 256 + threadIdx.x;
    if (idx >= W1PK_ELEMS) return;
    int h    = idx & 15;
    int lane = (idx >> 4) & 31;
    int tile = idx >> 9;                 // kt*NT_E + tn
    int tn = tile % NT_E;
    int kt = tile / NT_E;
    int k  = kt * 32 + (h & 7) + 8 * (lane >> 4) + 16 * (h >> 3);
    int n  = tn * 16 + (lane & 15);
    out[idx] = (k < HD_ && n < 200) ? (bf16)W1[k * 200 + n] : (bf16)0.0f;
}

// Embedding gather -> bf16, K padded 300 -> 320. One block per token row.
__global__ void gather_embed(const int* __restrict__ text, const float* __restrict__ emb,
                             bf16* __restrict__ xbf) {
    int row = blockIdx.x;                          // 0 .. B*L-1
    int tok = text[row];
    const float* e = emb + (size_t)tok * D_;
    bf16* dst = xbf + (size_t)row * DPAD;
    for (int c = threadIdx.x; c < DPAD; c += 256)
        dst[c] = (c < D_) ? (bf16)e[c] : (bf16)0.0f;
}

// M=64, N=400, K=320 bf16 WMMA GEMM: dst = Xs @ wp + bias.
// Called with the kernel-arg weight pointer directly so the backend keeps it in
// the global address space (global_load_b128 instead of flat_load_b128).
static __device__ __forceinline__ void qkv_gemm(const bf16* Xs,
                                                const bf16* __restrict__ wp,
                                                const float* __restrict__ bias,
                                                bf16* dst,
                                                int wv_, int lane, int lm, int g) {
    for (int t = wv_; t < 4 * NT_QKV; t += 8) {
        int tm = t / NT_QKV, tn = t % NT_QKV;
        int m = tm * 16 + lm;
        v8f acc = {0.f, 0.f, 0.f, 0.f, 0.f, 0.f, 0.f, 0.f};
        #pragma unroll
        for (int kt = 0; kt < KT_QKV; ++kt) {
            Frag a, bb;
            a.u[0] = *(const U4*)(Xs + m * DPAD + kt * 32 + 8 * g);
            a.u[1] = *(const U4*)(Xs + m * DPAD + kt * 32 + 16 + 8 * g);
            const U4* q0 = (const U4*)(wp + ((size_t)(kt * NT_QKV + tn) * 32 + lane) * 16);
            bb.u[0] = q0[0];
            bb.u[1] = q0[1];
            acc = wmma_bf16(a, bb, acc);
        }
        float bn = bias[tn * 16 + lm];
        #pragma unroll
        for (int r = 0; r < 8; ++r)
            dst[(tm * 16 + r + 8 * g) * HD_ + tn * 16 + lm] = (bf16)(acc[r] + bn);
    }
}

// Fully fused per-batch encoder: QKV -> exp-softmax attention -> additive pooling.
__global__ __launch_bounds__(256)
void fused_attn(const bf16* __restrict__ xbf,
                const bf16* __restrict__ wq, const bf16* __restrict__ wk, const bf16* __restrict__ wvp,
                const float* __restrict__ bq, const float* __restrict__ bk, const float* __restrict__ bv,
                const bf16* __restrict__ w1pk, const float* __restrict__ b1,
                const float* __restrict__ w2, const float* __restrict__ b2,
                float* __restrict__ out) {
    extern __shared__ char smem[];
    bf16*  Qs   = (bf16*)(smem + OFF_Q);
    bf16*  Ks   = (bf16*)(smem + OFF_K);
    bf16*  Vs   = (bf16*)(smem + OFF_V);
    bf16*  CTXs = (bf16*)(smem + OFF_CTX);
    float* Sf   = (float*)(smem + OFF_S);
    bf16*  At   = (bf16*)(smem + OFF_ATTN);
    float* Ef   = (float*)(smem + OFF_E);
    float* Al   = (float*)(smem + OFF_ALPHA);
    float* RSs  = (float*)(smem + OFF_RS);
    bf16*  Xs   = (bf16*)(smem + OFF_XS);

    const int tid  = threadIdx.x;
    const int wv_  = tid >> 5;     // wave id (wave32)
    const int lane = tid & 31;
    const int lm   = lane & 15;
    const int g    = lane >> 4;
    const int b    = blockIdx.x;

    // ---- stage x_b into LDS via the Tensor Data Mover (one wave issues the DMA) ----
#if defined(__has_builtin) && __has_builtin(__builtin_amdgcn_tensor_load_to_lds)
    if (wv_ == 0) {
        const uint32_t ldsb  = (uint32_t)(uintptr_t)smem + OFF_XS;
        const uint64_t gaddr = (uint64_t)(uintptr_t)(xbf + (size_t)b * L_ * DPAD);
        const uint32_t NU    = (uint32_t)(L_ * DPAD * 2 / 8);   // 5120 x 8-byte units
        // D# group0: count=1 | lds_addr | global_addr[56:0] | type=2
        v4u g0 = {1u, ldsb, (uint32_t)gaddr,
                  (uint32_t)((gaddr >> 32) & 0x1FFFFFFu) | (2u << 30)};
        // D# group1: data_size=3 (8B); tensor_dim0=tile_dim0=NU; tensor_dim1=tile_dim1=1;
        //            tensor_dim0_stride=NU
        v8i g1 = {(int)(3u << 16), (int)(NU << 16), (int)(1u << 16), (int)(NU << 16),
                  1, (int)NU, 0, 0};
        v4i gz = {0, 0, 0, 0};
#if __clang_major__ >= 23
        v8i gz8 = {0, 0, 0, 0, 0, 0, 0, 0};
        __builtin_amdgcn_tensor_load_to_lds(g0, g1, gz, gz, gz8, 0);
#else
        __builtin_amdgcn_tensor_load_to_lds(g0, g1, gz, gz, 0);
#endif
        __builtin_amdgcn_s_wait_tensorcnt(0);
    }
#else
    {
        const U4* src = (const U4*)(xbf + (size_t)b * L_ * DPAD);
        U4* dst = (U4*)Xs;
        for (int i = tid; i < L_ * DPAD / 8; i += 256) dst[i] = src[i];
    }
#endif
    // zero ctx K-padding columns + warm W1-pack lines for phase 3
    for (int i = tid; i < L_ * 16; i += 256)
        CTXs[(i >> 4) * CTXPAD + 400 + (i & 15)] = (bf16)0.0f;
    __builtin_prefetch(w1pk + (size_t)tid * (W1PK_ELEMS / 256), 0, 1);
    __syncthreads();

    // ---- Phase 1: Q/K/V = x @ W + b   (M=64, N=400, K=320, bf16 WMMA) ----
    qkv_gemm(Xs, wq,  bq, Qs, wv_, lane, lm, g);
    qkv_gemm(Xs, wk,  bk, Ks, wv_, lane, lm, g);
    qkv_gemm(Xs, wvp, bv, Vs, wv_, lane, lm, g);
    __syncthreads();

    // ---- Phase 2: per-head exp-softmax attention ----
    const float scale = 0.22360680f;   // 1/sqrt(20)
    for (int head = 0; head < H_; ++head) {
        // S = exp((Q_h K_h^T) * scale)   M=64, N=64, K=20 (padded 32)
        for (int t = wv_; t < 16; t += 8) {
            int tm = t >> 2, tn = t & 3;
            int m = tm * 16 + lm;
            int n = tn * 16 + lm;
            Frag a, bb;
            #pragma unroll
            for (int h = 0; h < 16; ++h) {
                int kk = h + 8 * g + ((h & 8) ? 8 : 0);   // 16-bit A layout
                a.v[h] = (kk < DK_) ? Qs[m * HD_ + head * DK_ + kk] : (bf16)0.0f;
            }
            #pragma unroll
            for (int h = 0; h < 16; ++h) {
                int kk = h + 8 * g + ((h & 8) ? 8 : 0);   // 16-bit B layout (B = K^T)
                bb.v[h] = (kk < DK_) ? Ks[n * HD_ + head * DK_ + kk] : (bf16)0.0f;
            }
            v8f acc = {0.f, 0.f, 0.f, 0.f, 0.f, 0.f, 0.f, 0.f};
            acc = wmma_bf16(a, bb, acc);
            #pragma unroll
            for (int r = 0; r < 8; ++r)
                Sf[(tm * 16 + r + 8 * g) * 64 + n] = __expf(acc[r] * scale);
        }
        __syncthreads();
        if (tid < 64) {
            float s = 1e-8f;
            const F4* pr = (const F4*)(Sf + tid * 64);
            #pragma unroll
            for (int k2 = 0; k2 < 16; ++k2) {
                F4 f = pr[k2];
                s += f.x + f.y + f.z + f.w;
            }
            RSs[tid] = 1.0f / s;
        }
        __syncthreads();
        for (int i = tid; i < 64 * 64; i += 256)
            At[i] = (bf16)(Sf[i] * RSs[i >> 6]);
        __syncthreads();
        // ctx_h = attn @ V_h   M=64, N=20 (padded 32), K=64
        {
            int t = wv_;                       // 8 tiles, one per wave
            int tm = t >> 1, tn = t & 1;
            int m = tm * 16 + lm;
            int n = tn * 16 + lm;
            v8f acc = {0.f, 0.f, 0.f, 0.f, 0.f, 0.f, 0.f, 0.f};
            #pragma unroll
            for (int kt = 0; kt < 2; ++kt) {
                Frag a, bb;
                a.u[0] = *(const U4*)(At + m * 64 + kt * 32 + 8 * g);
                a.u[1] = *(const U4*)(At + m * 64 + kt * 32 + 16 + 8 * g);
                #pragma unroll
                for (int h = 0; h < 16; ++h) {
                    int kr = kt * 32 + h + 8 * g + ((h & 8) ? 8 : 0);
                    bb.v[h] = (n < DK_) ? Vs[kr * HD_ + head * DK_ + n] : (bf16)0.0f;
                }
                acc = wmma_bf16(a, bb, acc);
            }
            if (n < DK_) {
                #pragma unroll
                for (int r = 0; r < 8; ++r)
                    CTXs[(tm * 16 + r + 8 * g) * CTXPAD + head * DK_ + n] = (bf16)acc[r];
            }
        }
        __syncthreads();
    }

    // ---- Phase 3: e = tanh(ctx @ W1 + b1)   M=64, N=208, K=416 ----
    for (int t = wv_; t < 4 * NT_E; t += 8) {
        int tm = t / NT_E, tn = t % NT_E;
        int m = tm * 16 + lm;
        int n = tn * 16 + lm;
        v8f acc = {0.f, 0.f, 0.f, 0.f, 0.f, 0.f, 0.f, 0.f};
        #pragma unroll
        for (int kt = 0; kt < KT_E; ++kt) {
            Frag a, bb;
            a.u[0] = *(const U4*)(CTXs + m * CTXPAD + kt * 32 + 8 * g);
            a.u[1] = *(const U4*)(CTXs + m * CTXPAD + kt * 32 + 16 + 8 * g);
            const U4* q0 = (const U4*)(w1pk + ((size_t)(kt * NT_E + tn) * 32 + lane) * 16);
            bb.u[0] = q0[0];
            bb.u[1] = q0[1];
            acc = wmma_bf16(a, bb, acc);
        }
        float bn = (n < 200) ? b1[n] : 0.0f;
        #pragma unroll
        for (int r = 0; r < 8; ++r)
            Ef[(tm * 16 + r + 8 * g) * NPADE + n] = tanhf(acc[r] + bn);
    }
    __syncthreads();

    // alpha = exp(e @ W2 + b2), normalize, weighted sum
    if (tid < 64) {
        float s = b2[0];
        const F4* pe = (const F4*)(Ef + tid * NPADE);
        const F4* pw = (const F4*)w2;
        #pragma unroll 5
        for (int j4 = 0; j4 < 50; ++j4) {
            F4 e4 = pe[j4];
            F4 w4 = pw[j4];
            s += e4.x * w4.x + e4.y * w4.y + e4.z * w4.z + e4.w * w4.w;
        }
        Al[tid] = __expf(s);
    }
    __syncthreads();
    if (tid == 0) {
        float s = 1e-8f;
        for (int l = 0; l < 64; ++l) s += Al[l];
        RSs[0] = 1.0f / s;
    }
    __syncthreads();
    float rn = RSs[0];
    for (int j = tid; j < HD_; j += 256) {
        float acco = 0.0f;
        for (int l = 0; l < 64; ++l)
            acco += (float)CTXs[l * CTXPAD + j] * Al[l];
        out[(size_t)b * HD_ + j] = acco * rn;
    }
}

extern "C" void kernel_launch(void* const* d_in, const int* in_sizes, int n_in,
                              void* d_out, int out_size, void* d_ws, size_t ws_size,
                              hipStream_t stream) {
    const int*   text = (const int*)d_in[0];
    const float* emb  = (const float*)d_in[1];
    const float* WQ   = (const float*)d_in[2];
    const float* bQ   = (const float*)d_in[3];
    const float* WK   = (const float*)d_in[4];
    const float* bK   = (const float*)d_in[5];
    const float* WV   = (const float*)d_in[6];
    const float* bV   = (const float*)d_in[7];
    const float* W1   = (const float*)d_in[8];
    const float* b1   = (const float*)d_in[9];
    const float* W2   = (const float*)d_in[10];
    const float* b2   = (const float*)d_in[11];
    float* out = (float*)d_out;

    char* ws   = (char*)d_ws;
    bf16* xbf  = (bf16*)ws;
    bf16* wqpk = (bf16*)(ws + XBF_BYTES);
    bf16* wkpk = wqpk + WPK_ELEMS;
    bf16* wvpk = wkpk + WPK_ELEMS;
    bf16* w1pk = wvpk + WPK_ELEMS;

    pack_qkv<<<(WPK_ELEMS + 255) / 256, 256, 0, stream>>>(WQ, wqpk);
    pack_qkv<<<(WPK_ELEMS + 255) / 256, 256, 0, stream>>>(WK, wkpk);
    pack_qkv<<<(WPK_ELEMS + 255) / 256, 256, 0, stream>>>(WV, wvpk);
    pack_w1<<<(W1PK_ELEMS + 255) / 256, 256, 0, stream>>>(W1, w1pk);
    gather_embed<<<B_ * L_, 256, 0, stream>>>(text, emb, xbf);
    fused_attn<<<B_, 256, LDS_BYTES, stream>>>(xbf, wqpk, wkpk, wvpk, bQ, bK, bV,
                                               w1pk, b1, W2, b2, out);
}
